// BiQRNNLayer_24790551232669
// MI455X (gfx1250) — matrix operand order repaired and checked by reading, not compiled, not run on hardware
//
#include <hip/hip_runtime.h>

typedef __attribute__((ext_vector_type(8)))  _Float16 v8h;
typedef __attribute__((ext_vector_type(16))) _Float16 v16h;
typedef __attribute__((ext_vector_type(8)))  float    v8f;

static constexpr int SEQ   = 2048;
static constexpr int BATCH = 16;
static constexpr int DIN   = 512;
static constexpr int HID   = 512;
static constexpr int NHB   = HID / 16;   // 32 hidden-unit blocks of 16

// ---------------------------------------------------------------------------
// Kernel 0: f32 -> f16 conversion (X and W staged once; both then live in L2)
// ---------------------------------------------------------------------------
__global__ void __launch_bounds__(256) cvt_f32_f16(const float* __restrict__ src,
                                                   _Float16* __restrict__ dst, int n) {
    int i = blockIdx.x * 256 + threadIdx.x;
    if (i < n) dst[i] = (_Float16)src[i];
}

// ---------------------------------------------------------------------------
// WMMA fragment loaders from row-major f16 [rows, ldk] tiles.
// A (16x32, MxK): lane l -> row l&15; elems 0..15 = K (l>>4)*16 + 0..15
// B (32x16, KxN): lane l -> col l&15; elems 0..15 = K (l>>4)*16 + 0..15
// ---------------------------------------------------------------------------
__device__ __forceinline__ v16h load_fragA(const _Float16* __restrict__ p0, int lane) {
    int r  = lane & 15;
    int kb = (lane >> 4) << 4;
    const _Float16* p = p0 + (size_t)r * DIN + kb;
    v8h lo = *(const v8h*)(p);
    v8h hi = *(const v8h*)(p + 8);
    return __builtin_shufflevector(lo, hi, 0,1,2,3,4,5,6,7,8,9,10,11,12,13,14,15);
}

__device__ __forceinline__ v16h load_fragB(const _Float16* __restrict__ p0, int lane) {
    int n  = lane & 15;
    int kb = (lane >> 4) << 4;
    const _Float16* p = p0 + (size_t)n * DIN + kb;
    v8h lo = *(const v8h*)(p);
    v8h hi = *(const v8h*)(p + 8);
    return __builtin_shufflevector(lo, hi, 0,1,2,3,4,5,6,7,8,9,10,11,12,13,14,15);
}

// ---------------------------------------------------------------------------
// Kernel 1: fused GEMM + bias + activations.
// One wave per (s, dir, h-block) task: computes z/f/o 16x16 tiles (shared A),
// stores a = 1-sigmoid(f), c = sigmoid(f)*tanh(z), so = sigmoid(o) as f16.
// task = s*64 + dir*32 + hb  (consecutive blocks share X[s] -> L2/L0 reuse)
// ---------------------------------------------------------------------------
__global__ void __launch_bounds__(256) qrnn_gemm(const _Float16* __restrict__ Xh,
                                                 const _Float16* __restrict__ Wh,
                                                 const float* __restrict__ b_fw,
                                                 const float* __restrict__ b_bw,
                                                 _Float16* __restrict__ Ag,
                                                 _Float16* __restrict__ Cg,
                                                 _Float16* __restrict__ Og) {
    const int lane = threadIdx.x & 31;
    const int wave = threadIdx.x >> 5;
    const unsigned task = blockIdx.x * 8u + wave;
    const int s   = task >> 6;
    const int dir = (task >> 5) & 1;
    const int hb  = task & 31;

    const _Float16* Xt = Xh + (size_t)s * (BATCH * DIN);          // 16 rows of X
    const _Float16* Wd = Wh + (size_t)dir * (3 * HID * DIN);
    const _Float16* Wz = Wd + (size_t)(0 * HID + hb * 16) * DIN;
    const _Float16* Wf = Wd + (size_t)(1 * HID + hb * 16) * DIN;
    const _Float16* Wo = Wd + (size_t)(2 * HID + hb * 16) * DIN;

    v8f accz = {}; v8f accf = {}; v8f acco = {};
    #pragma unroll 4
    for (int k0 = 0; k0 < DIN; k0 += 32) {
        v16h a  = load_fragA(Xt + k0, lane);
        v16h bz = load_fragB(Wz + k0, lane);
        v16h bf = load_fragB(Wf + k0, lane);
        v16h bo = load_fragB(Wo + k0, lane);
        accz = __builtin_amdgcn_wmma_f32_16x16x32_f16(false, a, false, bz, (short)0, accz, false, false);
        accf = __builtin_amdgcn_wmma_f32_16x16x32_f16(false, a, false, bf, (short)0, accf, false, false);
        acco = __builtin_amdgcn_wmma_f32_16x16x32_f16(false, a, false, bo, (short)0, acco, false, false);
    }

    const float* bias = dir ? b_bw : b_fw;
    const int n = lane & 15;
    const int h = hb * 16 + n;
    const float bz_ = bias[h];
    const float bf_ = bias[HID + h];
    const float bo_ = bias[2 * HID + h];
    const int mbase = (lane >> 4) << 3;
    const size_t outb = (((size_t)dir * SEQ + s) * BATCH) * HID;

    #pragma unroll
    for (int e = 0; e < 8; ++e) {
        const int b = mbase + e;                       // batch row (C/D layout)
        const float z = accz[e] + bz_;
        const float f = accf[e] + bf_;
        const float o = acco[e] + bo_;
        const float sf = 1.0f / (1.0f + __expf(-f));
        const float so = 1.0f / (1.0f + __expf(-o));
        const float tz = 1.0f - 2.0f / (__expf(2.0f * z) + 1.0f);   // tanh(z)
        const size_t idx = outb + (size_t)b * HID + h;
        Ag[idx] = (_Float16)(1.0f - sf);
        Cg[idx] = (_Float16)(sf * tz);
        Og[idx] = (_Float16)so;
    }
}

// ---------------------------------------------------------------------------
// Kernel 2: sequential fo-pool scan. One thread per (dir, b, h) channel,
// consecutive threads along h -> fully coalesced f16 gate reads / f32 writes.
// dir 0 scans forward; dir 1 scans backward (equivalent to flip/scan/flip).
// ---------------------------------------------------------------------------
__global__ void __launch_bounds__(256) qrnn_scan(const _Float16* __restrict__ Ag,
                                                 const _Float16* __restrict__ Cg,
                                                 const _Float16* __restrict__ Og,
                                                 float* __restrict__ out) {
    const int t   = blockIdx.x * 256 + threadIdx.x;   // 16384 channels
    const int dir = t >> 13;
    const int b   = (t >> 9) & 15;
    const int h   = t & 511;

    const size_t step  = (size_t)BATCH * HID;
    const size_t base  = (size_t)dir * SEQ * step + (size_t)b * HID + h;
    const size_t ostep = (size_t)BATCH * 2 * HID;
    const size_t obase = (size_t)b * (2 * HID) + (size_t)dir * HID + h;

    float hs = 0.0f;
    if (dir == 0) {
        for (int s = 0; s < SEQ; ++s) {
            const size_t i = base + (size_t)s * step;
            hs = (float)Ag[i] * hs + (float)Cg[i];
            out[obase + (size_t)s * ostep] = (float)Og[i] * hs;
        }
    } else {
        for (int s = SEQ - 1; s >= 0; --s) {
            const size_t i = base + (size_t)s * step;
            hs = (float)Ag[i] * hs + (float)Cg[i];
            out[obase + (size_t)s * ostep] = (float)Og[i] * hs;
        }
    }
}

// ---------------------------------------------------------------------------
// Launcher. Workspace layout (bytes):
//   [0, 33554432)                       Xh   f16  S*B*D
//   [33554432, 36700160)                Wh   f16  2 * 3H * D
//   [36700160, +64MB)  Ag               f16  2 * S*B*H
//   [... +64MB)        Cg
//   [... +64MB)        Og               total ~227 MB
// ---------------------------------------------------------------------------
extern "C" void kernel_launch(void* const* d_in, const int* in_sizes, int n_in,
                              void* d_out, int out_size, void* d_ws, size_t ws_size,
                              hipStream_t stream) {
    const float* X    = (const float*)d_in[0];
    const float* W_fw = (const float*)d_in[1];
    const float* b_fw = (const float*)d_in[2];
    const float* W_bw = (const float*)d_in[3];
    const float* b_bw = (const float*)d_in[4];
    float* out = (float*)d_out;

    char* ws = (char*)d_ws;
    _Float16* Xh = (_Float16*)(ws);
    _Float16* Wh = (_Float16*)(ws + 33554432);
    _Float16* Ag = (_Float16*)(ws + 36700160);
    _Float16* Cg = Ag + (size_t)2 * SEQ * BATCH * HID;
    _Float16* Og = Cg + (size_t)2 * SEQ * BATCH * HID;

    const int nX = SEQ * BATCH * DIN;   // 16,777,216
    const int nW = 3 * HID * DIN;       // 786,432
    cvt_f32_f16<<<nX / 256, 256, 0, stream>>>(X, Xh, nX);
    cvt_f32_f16<<<nW / 256, 256, 0, stream>>>(W_fw, Wh, nW);
    cvt_f32_f16<<<nW / 256, 256, 0, stream>>>(W_bw, Wh + nW, nW);

    // S * 2dir * 32 h-blocks = 131072 wave-tasks, 8 waves per 256-thread block
    qrnn_gemm<<<(SEQ * 2 * NHB) / 8, 256, 0, stream>>>(Xh, Wh, b_fw, b_bw, Ag, Cg, Og);

    // 16384 channels
    qrnn_scan<<<(2 * BATCH * HID) / 256, 256, 0, stream>>>(Ag, Cg, Og, out);
}
